// RotaryAttention_31851477467614
// MI455X (gfx1250) — matrix operand order repaired and checked by reading, not compiled, not run on hardware
//
#include <hip/hip_runtime.h>
#include <hip/hip_bf16.h>
#include <math.h>
#include <stdint.h>

#define D_MODEL   2048
#define NUM_HEADS 16
#define DK        128
#define SEQ       2048
#define BATCH     2
#define MTOT      (BATCH * SEQ)   // 4096 rows

typedef __attribute__((ext_vector_type(16))) _Float16 v16h;
typedef __attribute__((ext_vector_type(8)))  _Float16 v8h;
typedef __attribute__((ext_vector_type(8)))  float    v8f;

// ---------------------------------------------------------------------------
// WMMA wrapper: D = A(16x32 f16) * B(32x16 f16) + C(16x16 f32)
// ---------------------------------------------------------------------------
static __device__ inline v8f wmma_f16(v16h a, v16h b, v8f c) {
  return __builtin_amdgcn_wmma_f32_16x16x32_f16(
      /*neg_a=*/false, a, /*neg_b=*/false, b,
      /*c_mod=*/(short)0, c, /*reuse_a=*/false, /*reuse_b=*/false);
}

// Async DMA: 16 bytes global -> LDS per active lane, tracked with ASYNCcnt.
static __device__ inline void async_g2l_b128(unsigned lds_off, const void* gaddr) {
  asm volatile("global_load_async_to_lds_b128 %0, %1, off"
               :: "v"(lds_off), "v"((unsigned long long)(uintptr_t)gaddr)
               : "memory");
}
static __device__ inline void wait_async0() {
  asm volatile("s_wait_asynccnt 0x0" ::: "memory");
}

// A fragment (16x32, 16-bit): lane = M row (lane&15);
// elements 0..7  = K (lane>>4)*8 + 0..7 ; elements 8..15 = K 16 + (lane>>4)*8 + 0..7
static __device__ inline v16h ld_fragA(const _Float16* base, int ld, int row, int k0) {
  const int lane = threadIdx.x & 31;
  const _Float16* p = base + (size_t)(row + (lane & 15)) * ld + k0 + ((lane >> 4) << 3);
  v8h lo = *(const v8h*)p;
  v8h hi = *(const v8h*)(p + 16);
  v16h f;
#pragma unroll
  for (int i = 0; i < 8; ++i) { f[i] = lo[i]; f[i + 8] = hi[i]; }
  return f;
}

// B fragment (32x16, 16-bit) from an [N][K] row-major source (B = src^T):
// lane = N col (lane&15); elements 0..7 = K (lane>>4)*8 + 0..7 ; elements 8..15 = K 16 + (lane>>4)*8 + 0..7
static __device__ inline v16h ld_fragB(const _Float16* base, int ld, int col, int k0) {
  const int lane = threadIdx.x & 31;
  const _Float16* p = base + (size_t)(col + (lane & 15)) * ld + k0 + ((lane >> 4) << 3);
  v8h lo = *(const v8h*)p;
  v8h hi = *(const v8h*)(p + 16);
  v16h f;
#pragma unroll
  for (int i = 0; i < 8; ++i) { f[i] = lo[i]; f[i + 8] = hi[i]; }
  return f;
}

// ---------------------------------------------------------------------------
// fp32 -> f16 conversion
// ---------------------------------------------------------------------------
__global__ void f32_to_f16_kernel(const float* __restrict__ in,
                                  _Float16* __restrict__ out, size_t n) {
  size_t i = (size_t)blockIdx.x * blockDim.x + threadIdx.x;
  if (i < n) out[i] = (_Float16)in[i];
}

// ---------------------------------------------------------------------------
// GEMM: Y[M,N] = A[M,K] * W[N,K]^T + bias[N]
// Block = 128 threads = 4 waves (2x2), each wave owns a 64x64 tile:
// 16 WMMAs per k-step of 32 for 4 A-frag + 4 B-frag loads.
// PERM==1: f16 out in [B,H,S,DK];  PERM==0: f32 out in [M,N].
// ---------------------------------------------------------------------------
template <int PERM>
__global__ void __launch_bounds__(128)
gemm_kernel(const _Float16* __restrict__ A, const _Float16* __restrict__ W,
            const float* __restrict__ bias, void* __restrict__ out,
            int M, int Kdim, int N) {
  const int lane = threadIdx.x & 31;
  const int w    = threadIdx.x >> 5;
  const int r0   = blockIdx.y * 128 + (w >> 1) * 64;
  const int c0   = blockIdx.x * 128 + (w & 1) * 64;

  v8f acc[4][4];
#pragma unroll
  for (int i = 0; i < 4; ++i)
#pragma unroll
    for (int j = 0; j < 4; ++j)
#pragma unroll
      for (int e = 0; e < 8; ++e) acc[i][j][e] = 0.f;

  for (int k0 = 0; k0 < Kdim; k0 += 32) {
    if (k0 + 96 < Kdim) {   // prefetch 3 k-steps ahead (global_prefetch_b8)
      __builtin_prefetch(A + (size_t)(r0 + (lane & 15)) * Kdim + k0 + 96, 0, 0);
      __builtin_prefetch(W + (size_t)(c0 + (lane & 15)) * Kdim + k0 + 96, 0, 0);
    }
    v16h a[4], b[4];
#pragma unroll
    for (int i = 0; i < 4; ++i) a[i] = ld_fragA(A, Kdim, r0 + 16 * i, k0);
#pragma unroll
    for (int j = 0; j < 4; ++j) b[j] = ld_fragB(W, Kdim, c0 + 16 * j, k0);
#pragma unroll
    for (int i = 0; i < 4; ++i)
#pragma unroll
      for (int j = 0; j < 4; ++j) acc[i][j] = wmma_f16(a[i], b[j], acc[i][j]);
  }

  const int half = lane >> 4, l15 = lane & 15;
#pragma unroll
  for (int ti = 0; ti < 4; ++ti)
#pragma unroll
    for (int tj = 0; tj < 4; ++tj)
#pragma unroll
      for (int e = 0; e < 8; ++e) {
        int gr = r0 + 16 * ti + e + 8 * half;   // row in [0,M)
        int gc = c0 + 16 * tj + l15;            // col in [0,N)
        float v = acc[ti][tj][e] + bias[gc];
        if (PERM) {
          int b2 = gr >> 11, s = gr & (SEQ - 1);    // M = B*S, S=2048
          int h  = gc >> 7,  d = gc & (DK - 1);     // N = H*DK, DK=128
          ((_Float16*)out)[(((size_t)b2 * NUM_HEADS + h) * SEQ + s) * DK + d] =
              (_Float16)v;
        } else {
          ((float*)out)[(size_t)gr * N + gc] = v;
        }
      }
}

// ---------------------------------------------------------------------------
// RoPE in place on [B*H, S, DK] f16 (pairs (j, j+64)); optional scale folded in
// ---------------------------------------------------------------------------
__global__ void rope_kernel(_Float16* __restrict__ X, float scale) {
  size_t tid = (size_t)blockIdx.x * blockDim.x + threadIdx.x;
  size_t total = (size_t)BATCH * NUM_HEADS * SEQ * (DK / 2);
  if (tid >= total) return;
  int j = tid & 63;
  size_t rest = tid >> 6;
  int s = (int)(rest & (SEQ - 1));
  size_t bh = rest >> 11;
  _Float16* p = X + (bh * SEQ + s) * (size_t)DK;
  float inv_freq = powf(10000.f, -(float)j / 64.f);
  float ang = (float)s * inv_freq;
  float si, co;
  sincosf(ang, &si, &co);
  float x1 = (float)p[j];
  float x2 = (float)p[j + 64];
  p[j]      = (_Float16)((x1 * co - x2 * si) * scale);
  p[j + 64] = (_Float16)((x2 * co + x1 * si) * scale);
}

// ---------------------------------------------------------------------------
// Flash attention: grid(B*H, S/64), block 128 (4 waves, 16 q rows each).
// K/V tiles staged into LDS with async DMA (ASYNCcnt), double-buffered so the
// copy of tile t+1 overlaps WMMA work on tile t. Score scale pre-folded into Q.
// ---------------------------------------------------------------------------
__global__ void __launch_bounds__(128)
flash_attn_kernel(const _Float16* __restrict__ Q, const _Float16* __restrict__ K,
                  const _Float16* __restrict__ V, _Float16* __restrict__ Out) {
  __shared__ _Float16 Ks[2][32][136];   // [buf][key][d], row stride 272B (16B mult)
  __shared__ _Float16 Vs[2][32][136];   // [buf][key][d]
  __shared__ _Float16 Pb[4][16][32];    // per-wave P staging (C-layout -> A-layout)

  const int lane = threadIdx.x & 31;
  const int w    = threadIdx.x >> 5;
  const int half = lane >> 4;
  const int l15  = lane & 15;
  const int bh   = blockIdx.x;
  const int q0   = blockIdx.y * 64;

  const _Float16* Qp = Q + (size_t)bh * SEQ * DK;
  const _Float16* Kp = K + (size_t)bh * SEQ * DK;
  const _Float16* Vp = V + (size_t)bh * SEQ * DK;

  // Async-stage one 32-key K/V tile pair into buffer `buf`
  auto stage = [&](int kv, int buf) {
    for (int idx = threadIdx.x; idx < 512; idx += 128) {
      int r  = idx >> 4;            // key 0..31
      int cc = (idx & 15) * 8;      // d chunk (8 halfs = 16B)
      async_g2l_b128((unsigned)(uintptr_t)&Ks[buf][r][cc],
                     Kp + (size_t)(kv + r) * DK + cc);
      async_g2l_b128((unsigned)(uintptr_t)&Vs[buf][r][cc],
                     Vp + (size_t)(kv + r) * DK + cc);
    }
  };

  // Preload this wave's 16 query rows as 4 A-fragments (K-chunks of 32)
  v16h qa[4];
#pragma unroll
  for (int kc = 0; kc < 4; ++kc) qa[kc] = ld_fragA(Qp, DK, q0 + w * 16, kc * 32);

  v8f o[8];
#pragma unroll
  for (int f = 0; f < 8; ++f)
#pragma unroll
    for (int e = 0; e < 8; ++e) o[f][e] = 0.f;
  float mrow[8], lrow[8];
#pragma unroll
  for (int i = 0; i < 8; ++i) { mrow[i] = -1e30f; lrow[i] = 0.f; }

  stage(0, 0);
  wait_async0();
  __syncthreads();

  const int niter = SEQ / 32;
  for (int it = 0; it < niter; ++it) {
    const int buf = it & 1;
    if (it + 1 < niter) stage((it + 1) * 32, buf ^ 1);   // overlap DMA w/ compute

    const _Float16* Kb = &Ks[buf][0][0];

    // --- scores: S(16x32) = Q(16x128) * Ktile^T, two 16x16 WMMA chains ---
    v8f s0, s1;
#pragma unroll
    for (int e = 0; e < 8; ++e) { s0[e] = 0.f; s1[e] = 0.f; }
    {
      v16h kb[4];
#pragma unroll
      for (int kc = 0; kc < 4; ++kc) kb[kc] = ld_fragB(Kb, 136, 0, kc * 32);
#pragma unroll
      for (int kc = 0; kc < 4; ++kc) s0 = wmma_f16(qa[kc], kb[kc], s0);
#pragma unroll
      for (int kc = 0; kc < 4; ++kc) kb[kc] = ld_fragB(Kb, 136, 16, kc * 32);
#pragma unroll
      for (int kc = 0; kc < 4; ++kc) s1 = wmma_f16(qa[kc], kb[kc], s1);
    }

    // --- online softmax (row stats reduced across the 16-lane C groups) ---
    float alpha[8], p0[8], p1[8];
#pragma unroll
    for (int i = 0; i < 8; ++i) {
      float t = fmaxf(s0[i], s1[i]);
      t = fmaxf(t, __shfl_xor(t, 1, 32));
      t = fmaxf(t, __shfl_xor(t, 2, 32));
      t = fmaxf(t, __shfl_xor(t, 4, 32));
      t = fmaxf(t, __shfl_xor(t, 8, 32));
      float mnew = fmaxf(mrow[i], t);
      alpha[i] = __expf(mrow[i] - mnew);
      p0[i] = __expf(s0[i] - mnew);
      p1[i] = __expf(s1[i] - mnew);
      float r = p0[i] + p1[i];
      r += __shfl_xor(r, 1, 32);
      r += __shfl_xor(r, 2, 32);
      r += __shfl_xor(r, 4, 32);
      r += __shfl_xor(r, 8, 32);
      lrow[i] = lrow[i] * alpha[i] + r;
      mrow[i] = mnew;
    }
#pragma unroll
    for (int f = 0; f < 8; ++f)
#pragma unroll
      for (int i = 0; i < 8; ++i) o[f][i] = o[f][i] * alpha[i];

    // --- P: C-layout -> LDS -> A-layout fragment ---
#pragma unroll
    for (int i = 0; i < 8; ++i) {
      Pb[w][i + 8 * half][l15]      = (_Float16)p0[i];
      Pb[w][i + 8 * half][16 + l15] = (_Float16)p1[i];
    }
    v16h pa;
    {
      const _Float16* p = &Pb[w][l15][0];
      v8h lo = *(const v8h*)(p + half * 8);
      v8h hi = *(const v8h*)(p + 16 + half * 8);
#pragma unroll
      for (int i = 0; i < 8; ++i) { pa[i] = lo[i]; pa[i + 8] = hi[i]; }
    }

    // --- O(16x128) += P(16x32) * Vtile(32x128); B-frags gathered from LDS ---
#pragma unroll
    for (int f = 0; f < 8; ++f) {
      v16h vb;
#pragma unroll
      for (int j = 0; j < 16; ++j)
        vb[j] = Vs[buf][(j & 7) + half * 8 + (j >> 3) * 16][16 * f + l15];   // element j = key (table order)
      o[f] = wmma_f16(pa, vb, o[f]);
    }

    wait_async0();       // my DMA for the other buffer is done
    __syncthreads();     // everyone finished reading buf / writing buf^1
  }

  // --- normalize + write to [B, S, D_MODEL] f16 ---
  const int b = bh / NUM_HEADS, h = bh % NUM_HEADS;
#pragma unroll
  for (int i = 0; i < 8; ++i) {
    float inv = (lrow[i] > 0.f) ? 1.f / lrow[i] : 0.f;
    int gr = q0 + w * 16 + i + 8 * half;
    size_t rowbase = ((size_t)b * SEQ + gr) * D_MODEL + (size_t)h * DK;
#pragma unroll
    for (int f = 0; f < 8; ++f)
      Out[rowbase + 16 * f + l15] = (_Float16)(o[f][i] * inv);
  }
}

// ---------------------------------------------------------------------------
// Host launcher
// ---------------------------------------------------------------------------
extern "C" void kernel_launch(void* const* d_in, const int* in_sizes, int n_in,
                              void* d_out, int out_size, void* d_ws, size_t ws_size,
                              hipStream_t stream) {
  const float* x  = (const float*)d_in[0];
  const float* wq = (const float*)d_in[1];
  const float* bq = (const float*)d_in[2];
  const float* wk = (const float*)d_in[3];
  const float* bk = (const float*)d_in[4];
  const float* wv = (const float*)d_in[5];
  const float* bv = (const float*)d_in[6];
  const float* wo = (const float*)d_in[7];
  const float* bo = (const float*)d_in[8];

  const size_t NE_X = (size_t)MTOT * D_MODEL;
  const size_t NE_W = (size_t)D_MODEL * D_MODEL;
  const size_t NE_H = (size_t)BATCH * NUM_HEADS * SEQ * DK;

  _Float16* p   = (_Float16*)d_ws;
  _Float16* xh  = p; p += NE_X;
  _Float16* wqh = p; p += NE_W;
  _Float16* wkh = p; p += NE_W;
  _Float16* wvh = p; p += NE_W;
  _Float16* woh = p; p += NE_W;
  _Float16* Qh  = p; p += NE_H;
  _Float16* Kh  = p; p += NE_H;
  _Float16* Vh  = p; p += NE_H;
  _Float16* Ah  = p; p += NE_X;

  auto cvt = [&](const float* src, _Float16* dst, size_t n) {
    f32_to_f16_kernel<<<(unsigned)((n + 255) / 256), 256, 0, stream>>>(src, dst, n);
  };
  cvt(x,  xh,  NE_X);
  cvt(wq, wqh, NE_W);
  cvt(wk, wkh, NE_W);
  cvt(wv, wvh, NE_W);
  cvt(wo, woh, NE_W);

  dim3 gg(D_MODEL / 128, MTOT / 128);   // (16, 32)
  gemm_kernel<1><<<gg, 128, 0, stream>>>(xh, wqh, bq, Qh, MTOT, D_MODEL, D_MODEL);
  gemm_kernel<1><<<gg, 128, 0, stream>>>(xh, wkh, bk, Kh, MTOT, D_MODEL, D_MODEL);
  gemm_kernel<1><<<gg, 128, 0, stream>>>(xh, wvh, bv, Vh, MTOT, D_MODEL, D_MODEL);

  const size_t rope_n = (size_t)BATCH * NUM_HEADS * SEQ * (DK / 2);
  rope_kernel<<<(unsigned)((rope_n + 255) / 256), 256, 0, stream>>>(
      Qh, 0.08838834764831845f /* 1/sqrt(128) */);
  rope_kernel<<<(unsigned)((rope_n + 255) / 256), 256, 0, stream>>>(Kh, 1.0f);

  flash_attn_kernel<<<dim3(BATCH * NUM_HEADS, SEQ / 64), 128, 0, stream>>>(
      Qh, Kh, Vh, Ah);

  gemm_kernel<0><<<gg, 128, 0, stream>>>(Ah, woh, bo, d_out, MTOT, D_MODEL, D_MODEL);
}
